// Featurizer_36146444763193
// MI455X (gfx1250) — matrix operand order repaired and checked by reading, not compiled, not run on hardware
//
#include <hip/hip_runtime.h>
#include <hip/hip_bf16.h>
#include <stdint.h>

// ---------------------------------------------------------------------------
// Problem constants (from reference)
// ---------------------------------------------------------------------------
#define NRES   8192
#define NEDGE  196608
#define NA     6            // atoms per residue
#define HDIM   128
#define CENTRAL_ATOM 3
#define EPSV   1e-06f

#define KE     688          // edge feature length
#define KE_PAD 704          // 22 chunks of 32
#define KE_CHUNKS 22
#define KN     44           // node feature length
#define KN_PAD 64           // 2 chunks of 32
#define KN_CHUNKS 2
#define NTILES 8            // 128 / 16

// RBF parameters: mu = linspace(0,20,16) -> spacing 20/15 ; sigma = 20/16
#define RBF_MU_STEP  (20.0f / 15.0f)
#define RBF_INV_SIG  (16.0f / 20.0f)

// workspace layout (bytes)
#define Q_OFF   0                       // 8192*9 f32  = 294912
#define QV_OFF  294912                  // 8192 int    = 32768
#define WE_OFF  327680                  // 704*128 f16 = 180224
#define WN_OFF  507904                  // 64*128 f16  = 16384
#define HV_OFF  524288                  // 8192*64 f16 = 1048576
// total: 1572864 bytes

typedef __attribute__((ext_vector_type(16))) _Float16 v16h;
typedef __attribute__((ext_vector_type(8)))  _Float16 v8h;
typedef __attribute__((ext_vector_type(4)))  _Float16 v4h;
typedef __attribute__((ext_vector_type(8)))  float    v8f;

__device__ __forceinline__ float sgnf(float x) {
    return (x > 0.0f) ? 1.0f : ((x < 0.0f) ? -1.0f : 0.0f);
}

// ---------------------------------------------------------------------------
// Pack w_edge (688x128 f32) and w_node (44x128 f32) into per-lane f16
// B-fragments for v_wmma_f32_16x16x32_f16.
// B layout (32x16 f16): column N = lane&15 ; K = (lane>=16 ? 8 : 0) + (half&7) + (half>=8 ? 16 : 0).
// Fragment for (chunk c, ntile t): lane l holds 16 contiguous halves at
//   packed[ ((c*8 + t)*32 + l)*16 + h ]
// ---------------------------------------------------------------------------
__global__ __launch_bounds__(256) void pack_w_kernel(const float* __restrict__ w_edge,
                                                     const float* __restrict__ w_node,
                                                     _Float16* __restrict__ wE,
                                                     _Float16* __restrict__ wN) {
    const int idx = blockIdx.x * 256 + threadIdx.x;
    const int NE_ELEMS = KE_CHUNKS * NTILES * 32 * 16;   // 90112
    const int NN_ELEMS = KN_CHUNKS * NTILES * 32 * 16;   // 8192
    if (idx < NE_ELEMS) {
        const int h = idx & 15, lane = (idx >> 4) & 31, t = (idx >> 9) & 7, c = idx >> 12;
        const int k = c * 32 + ((lane >> 4) << 3) + (h & 7) + ((h >> 3) << 4);
        const int n = t * 16 + (lane & 15);
        wE[idx] = (k < KE) ? (_Float16)w_edge[k * HDIM + n] : (_Float16)0.0f;
    } else if (idx < NE_ELEMS + NN_ELEMS) {
        const int u = idx - NE_ELEMS;
        const int h = u & 15, lane = (u >> 4) & 31, t = (u >> 9) & 7, c = u >> 12;
        const int k = c * 32 + ((lane >> 4) << 3) + (h & 7) + ((h >> 3) << 4);
        const int n = t * 16 + (lane & 15);
        wN[u] = (k < KN) ? (_Float16)w_node[k * HDIM + n] : (_Float16)0.0f;
    }
}

// ---------------------------------------------------------------------------
// Per-residue local frame Q (3x3, stored row-major Q[j][i] at q[i*9 + j*3 + i])
// zeroed where qv==0, plus qv flags.
// ---------------------------------------------------------------------------
__global__ __launch_bounds__(256) void prep_q_kernel(const float* __restrict__ central,
                                                     const int* __restrict__ batch,
                                                     const int* __restrict__ chain,
                                                     float* __restrict__ Qout,
                                                     int* __restrict__ qv) {
    const int i = blockIdx.x * 256 + threadIdx.x;
    if (i >= NRES) return;
    const bool chg  = (i >= 1)        && (batch[i] != batch[i-1] || chain[i] != chain[i-1]);
    const bool chgn = (i < NRES - 1)  && (batch[i+1] != batch[i] || chain[i+1] != chain[i]);
    const bool valid = (i >= 1) && (i <= NRES - 2) && !chg && !chgn;
    qv[i] = valid ? 1 : 0;
    float q[9];
    for (int j = 0; j < 9; ++j) q[j] = 0.0f;
    if (valid) {
        float u0[3], u1[3];
        float n0 = 0.0f, n1 = 0.0f;
        for (int j = 0; j < 3; ++j) {
            u0[j] = central[i*3 + j]     - central[(i-1)*3 + j];
            u1[j] = central[(i+1)*3 + j] - central[i*3 + j];
            n0 += u0[j]*u0[j]; n1 += u1[j]*u1[j];
        }
        n0 = 1.0f / fmaxf(sqrtf(n0), 1e-12f);
        n1 = 1.0f / fmaxf(sqrtf(n1), 1e-12f);
        for (int j = 0; j < 3; ++j) { u0[j] *= n0; u1[j] *= n1; }
        float b[3] = {u0[0]-u1[0], u0[1]-u1[1], u0[2]-u1[2]};
        float nv[3] = {u0[1]*u1[2]-u0[2]*u1[1], u0[2]*u1[0]-u0[0]*u1[2], u0[0]*u1[1]-u0[1]*u1[0]};
        float nb = 1.0f / fmaxf(sqrtf(b[0]*b[0]+b[1]*b[1]+b[2]*b[2]), 1e-12f);
        float nn = 1.0f / fmaxf(sqrtf(nv[0]*nv[0]+nv[1]*nv[1]+nv[2]*nv[2]), 1e-12f);
        for (int j = 0; j < 3; ++j) { b[j] *= nb; nv[j] *= nn; }
        float c3[3] = {b[1]*nv[2]-b[2]*nv[1], b[2]*nv[0]-b[0]*nv[2], b[0]*nv[1]-b[1]*nv[0]};
        // Q[j][col]: col0=b, col1=nv, col2=b x nv
        for (int j = 0; j < 3; ++j) { q[j*3+0] = b[j]; q[j*3+1] = nv[j]; q[j*3+2] = c3[j]; }
    }
    for (int j = 0; j < 9; ++j) Qout[i*9 + j] = q[j];
}

// ---------------------------------------------------------------------------
// Node features (44) per residue -> f16 rows of width 64 (zero padded).
// ---------------------------------------------------------------------------
__device__ __forceinline__ void norm3(float* v) {
    float n = 1.0f / fmaxf(sqrtf(v[0]*v[0]+v[1]*v[1]+v[2]*v[2]), 1e-12f);
    v[0] *= n; v[1] *= n; v[2] *= n;
}

__global__ __launch_bounds__(256) void node_feat_kernel(const float* __restrict__ coords,
                                                        const float* __restrict__ central,
                                                        const int* __restrict__ batch,
                                                        const int* __restrict__ chain,
                                                        const float* __restrict__ Q,
                                                        _Float16* __restrict__ hV) {
    const int i = blockIdx.x * 256 + threadIdx.x;
    if (i >= NRES) return;
    const int N = NRES * NA;
    const bool chg  = (i >= 1)       && (batch[i] != batch[i-1] || chain[i] != chain[i-1]);
    const bool chgn = (i < NRES - 1) && (batch[i+1] != batch[i] || chain[i+1] != chain[i]);

    float feat[KN];
    for (int f = 0; f < KN; ++f) feat[f] = 0.0f;

    for (int a = 0; a < NA; ++a) {
        const int k = i * NA + a;
        // dihedral: valid flat range [1, N-3]
        bool dvok = (k >= 1) && (k <= N - 3) && !(chg && a == 0) && !(chgn && a >= NA - 2);
        if (dvok) {
            float U0[3], U1[3], U2[3];
            for (int j = 0; j < 3; ++j) {
                U0[j] = coords[k*3 + j]     - coords[(k-1)*3 + j];
                U1[j] = coords[(k+1)*3 + j] - coords[k*3 + j];
                U2[j] = coords[(k+2)*3 + j] - coords[(k+1)*3 + j];
            }
            norm3(U0); norm3(U1); norm3(U2);
            float c1[3] = {U0[1]*U1[2]-U0[2]*U1[1], U0[2]*U1[0]-U0[0]*U1[2], U0[0]*U1[1]-U0[1]*U1[0]};
            float c2[3] = {U1[1]*U2[2]-U1[2]*U2[1], U1[2]*U2[0]-U1[0]*U2[2], U1[0]*U2[1]-U1[1]*U2[0]};
            norm3(c1); norm3(c2);
            float cd = c1[0]*c2[0]+c1[1]*c2[1]+c1[2]*c2[2];
            cd = fminf(fmaxf(cd, -1.0f + EPSV), 1.0f - EPSV);
            float d = acosf(cd) * sgnf(c2[0]*U0[0]+c2[1]*U0[1]+c2[2]*U0[2]);
            feat[2*a]   = sinf(d);
            feat[2*a+1] = cosf(d);
        }
        // angle: valid flat range [1, N-2]
        bool avok = (k >= 1) && (k <= N - 2) && !(chg && a == 0) && !(chgn && a == NA - 1);
        if (avok) {
            float d0[3], d1[3];
            for (int j = 0; j < 3; ++j) {
                d0[j] = coords[(k-1)*3 + j] - coords[k*3 + j];
                d1[j] = coords[(k+1)*3 + j] - coords[k*3 + j];
            }
            norm3(d0); norm3(d1);
            float cv = d0[0]*d1[0]+d0[1]*d1[1]+d0[2]*d1[2];
            float sv = sqrtf(1.0f - cv*cv + EPSV);
            feat[12 + 2*a]   = sv;
            feat[12 + 2*a+1] = cv;
        }
    }
    // intra_dist (5) + intra_direct (15)
    float Qi[9];
    for (int j = 0; j < 9; ++j) Qi[j] = Q[i*9 + j];
    int qidx = 0;
    for (int a = 0; a < NA; ++a) {
        if (a == CENTRAL_ATOM) continue;
        float v[3];
        float d2 = 0.0f;
        for (int j = 0; j < 3; ++j) {
            v[j] = coords[(i*NA + a)*3 + j] - central[i*3 + j];
            d2 += v[j]*v[j];
        }
        feat[24 + qidx] = logf(sqrtf(d2 + EPSV));
        float inv = 1.0f / fmaxf(sqrtf(d2), 1e-12f);
        float dn[3] = {v[0]*inv, v[1]*inv, v[2]*inv};
        for (int ii = 0; ii < 3; ++ii)
            feat[29 + qidx*3 + ii] = Qi[0*3+ii]*dn[0] + Qi[1*3+ii]*dn[1] + Qi[2*3+ii]*dn[2];
        ++qidx;
    }
    _Float16* row = hV + (size_t)i * KN_PAD;
    for (int f = 0; f < KN; ++f)   row[f] = (_Float16)feat[f];
    for (int f = KN; f < KN_PAD; ++f) row[f] = (_Float16)0.0f;
}

// ---------------------------------------------------------------------------
// WMMA fragment conventions (see CDNA5 ISA §7.12.2):
// A (16x32 f16): lanes 0-15 -> M=lane, K in {0..7, 16..23};
//                lanes 16-31 -> M=lane-16, K in {8..15, 24..31}
//   => per lane: two contiguous 16B LDS reads at K offsets {base, base+16},
//      base = chunk*32 + (lane>=16 ? 8 : 0)
// C/D: VGPR r, lane l: M = (l<16 ? r : r+8), N = tile*16 + (l&15).
// ---------------------------------------------------------------------------

// ---------------- Node GEMM (M-tiles of 16 residues) ------------------------
__global__ __launch_bounds__(128) void node_gemm_kernel(const _Float16* __restrict__ hV,
                                                        const _Float16* __restrict__ wN,
                                                        const float* __restrict__ bN,
                                                        const float* __restrict__ gN,
                                                        const float* __restrict__ beN,
                                                        float* __restrict__ out) {
    __shared__ __align__(16) _Float16 featN[16 * KN_PAD];
    __shared__ float outT[16][HDIM];
    __shared__ float psum[16][8], psq[16][8];
    __shared__ float mrow[16], rrow[16];

    const int tid = threadIdx.x;
    // copy 16x64 f16 tile (1024 halves) -> 8 halves per thread
    ((v8h*)featN)[tid] = ((const v8h*)(hV + (size_t)blockIdx.x * 16 * KN_PAD))[tid];
    __syncthreads();

    const int lane = tid & 31, wave = tid >> 5;
    const int row = lane & 15, hi = lane >> 4;
    const int t0 = wave * 2, t1 = t0 + 1;
    v8f acc0 = {}, acc1 = {};
    for (int c = 0; c < KN_CHUNKS; ++c) {
        union { v16h v; v8h h2[2]; } Af, B0, B1;
        const _Float16* ap = &featN[row * KN_PAD + c * 32 + hi * 8];
        Af.h2[0] = *(const v8h*)ap;
        Af.h2[1] = *(const v8h*)(ap + 16);
        const v8h* bp0 = (const v8h*)(wN + (size_t)((c * NTILES + t0) * 32 + lane) * 16);
        const v8h* bp1 = (const v8h*)(wN + (size_t)((c * NTILES + t1) * 32 + lane) * 16);
        B0.h2[0] = bp0[0]; B0.h2[1] = bp0[1];
        B1.h2[0] = bp1[0]; B1.h2[1] = bp1[1];
        acc0 = __builtin_amdgcn_wmma_f32_16x16x32_f16(false, Af.v, false, B0.v, (short)0, acc0, false, false);
        acc1 = __builtin_amdgcn_wmma_f32_16x16x32_f16(false, Af.v, false, B1.v, (short)0, acc1, false, false);
    }
    for (int r = 0; r < 8; ++r) {
        const int rr = hi ? (r + 8) : r;
        const int n0 = t0 * 16 + row, n1 = t1 * 16 + row;
        outT[rr][n0] = acc0[r] + bN[n0];
        outT[rr][n1] = acc1[r] + bN[n1];
    }
    __syncthreads();
    // layernorm over H=128 per row
    const int lrow = tid >> 3, seg = tid & 7;
    float s1 = 0.0f, s2 = 0.0f;
    for (int j = 0; j < 16; ++j) { float x = outT[lrow][seg*16 + j]; s1 += x; s2 += x*x; }
    psum[lrow][seg] = s1; psq[lrow][seg] = s2;
    __syncthreads();
    if (tid < 16) {
        float S1 = 0.0f, S2 = 0.0f;
        for (int j = 0; j < 8; ++j) { S1 += psum[tid][j]; S2 += psq[tid][j]; }
        float mean = S1 / (float)HDIM;
        float var  = S2 / (float)HDIM - mean * mean;
        mrow[tid] = mean;
        rrow[tid] = rsqrtf(var + 1e-05f);
    }
    __syncthreads();
    const float mean = mrow[lrow], rstd = rrow[lrow];
    const size_t res = (size_t)blockIdx.x * 16 + lrow;
    for (int j = 0; j < 16; ++j) {
        const int n = seg * 16 + j;
        const float x = outT[lrow][n];
        out[res * HDIM + n] = gN[n] * (x - mean) * rstd + beN[n];
    }
}

// ---------------- Fused edge featurize + GEMM + LN --------------------------
__global__ __launch_bounds__(128) void edge_kernel(const float* __restrict__ coords,
                                                   const int* __restrict__ edge_index,
                                                   const float* __restrict__ Q,
                                                   const int* __restrict__ qv,
                                                   const _Float16* __restrict__ wE,
                                                   const float* __restrict__ bE,
                                                   const float* __restrict__ gE,
                                                   const float* __restrict__ beE,
                                                   float* __restrict__ outE) {
    __shared__ __align__(16) _Float16 feat[16 * KE_PAD];   // 22528 B
    __shared__ float outT[16][HDIM];                       // 8192 B
    __shared__ float psum[16][8], psq[16][8];
    __shared__ float mrow[16], rrow[16];

    const int tid  = threadIdx.x;
    const int slot = tid >> 3;          // edge slot 0..15
    const int sub  = tid & 7;           // 8 threads per edge
    const int e    = blockIdx.x * 16 + slot;
    const int s    = edge_index[e];
    const int t    = edge_index[NEDGE + e];

    float Qt[9];
    for (int j = 0; j < 9; ++j) Qt[j] = Q[t*9 + j];

    // 36 atom pairs: diff = coords[src][n] - coords[tgt][m]
    for (int p = sub; p < 36; p += 8) {
        const int m = p / 6, n = p % 6;
        const float dx = coords[(s*NA + n)*3 + 0] - coords[(t*NA + m)*3 + 0];
        const float dy = coords[(s*NA + n)*3 + 1] - coords[(t*NA + m)*3 + 1];
        const float dz = coords[(s*NA + n)*3 + 2] - coords[(t*NA + m)*3 + 2];
        const float d2 = dx*dx + dy*dy + dz*dz;
        const float Din = sqrtf(d2 + EPSV);
        // 16 RBF values -> registers, then 4 x ds_store_b64
        union { _Float16 h[16]; v4h q[4]; } rbf;
        #pragma unroll
        for (int r = 0; r < 16; ++r) {
            const float u = (Din - RBF_MU_STEP * (float)r) * RBF_INV_SIG;
            rbf.h[r] = (_Float16)__expf(-u * u);
        }
        v4h* fp = (v4h*)&feat[slot * KE_PAD + 4 + p * 16];   // byte off 8+32p: 8B aligned
        #pragma unroll
        for (int qq = 0; qq < 4; ++qq) fp[qq] = rbf.q[qq];

        const float inv = 1.0f / fmaxf(sqrtf(d2), 1e-12f);
        const float vx = dx*inv, vy = dy*inv, vz = dz*inv;
        _Float16* dp = &feat[slot * KE_PAD + 580 + p * 3];
        #pragma unroll
        for (int ii = 0; ii < 3; ++ii)
            dp[ii] = (_Float16)(Qt[0*3+ii]*vx + Qt[1*3+ii]*vy + Qt[2*3+ii]*vz);
    }
    if (sub == 0) {
        // orientation quaternion of Rm = Q[tgt]^T * Q[src]
        float Qs[9];
        for (int j = 0; j < 9; ++j) Qs[j] = Q[s*9 + j];
        float R[3][3];
        for (int ii = 0; ii < 3; ++ii)
            for (int kk = 0; kk < 3; ++kk)
                R[ii][kk] = Qt[0*3+ii]*Qs[0*3+kk] + Qt[1*3+ii]*Qs[1*3+kk] + Qt[2*3+ii]*Qs[2*3+kk];
        const float Rxx = R[0][0], Ryy = R[1][1], Rzz = R[2][2];
        const float m0 = 0.5f * sqrtf(fabsf(1.0f + Rxx - Ryy - Rzz));
        const float m1 = 0.5f * sqrtf(fabsf(1.0f - Rxx + Ryy - Rzz));
        const float m2 = 0.5f * sqrtf(fabsf(1.0f - Rxx - Ryy + Rzz));
        const float q0 = sgnf(R[2][1] - R[1][2]) * m0;
        const float q1 = sgnf(R[0][2] - R[2][0]) * m1;
        const float q2 = sgnf(R[1][0] - R[0][1]) * m2;
        const float q3 = 0.5f * sqrtf(fmaxf(1.0f + Rxx + Ryy + Rzz, 0.0f));
        const float invn = 1.0f / fmaxf(sqrtf(q0*q0 + q1*q1 + q2*q2 + q3*q3), 1e-12f);
        const float ev = (qv[s] && qv[t]) ? invn : 0.0f;
        union { _Float16 h[4]; v4h q; } ou;
        ou.h[0] = (_Float16)(q0 * ev);
        ou.h[1] = (_Float16)(q1 * ev);
        ou.h[2] = (_Float16)(q2 * ev);
        ou.h[3] = (_Float16)(q3 * ev);
        *(v4h*)&feat[slot*KE_PAD + 0] = ou.q;                // byte off 0: aligned
    }
    if (sub == 1) {
        // zero pad K = 688..703 (16 halves, 8B aligned at byte off 1376)
        v4h z = {};
        v4h* zp = (v4h*)&feat[slot*KE_PAD + KE];
        zp[0] = z; zp[1] = z; zp[2] = z; zp[3] = z;
    }
    __syncthreads();

    // ---- WMMA GEMM: 16 edges x 128 channels, K = 704 -----------------------
    const int lane = tid & 31, wave = tid >> 5;
    const int row = lane & 15, hi = lane >> 4;
    const int t0 = wave * 2, t1 = t0 + 1;
    v8f acc0 = {}, acc1 = {};
    for (int c = 0; c < KE_CHUNKS; ++c) {
        union { v16h v; v8h h2[2]; } Af, B0, B1;
        const _Float16* ap = &feat[row * KE_PAD + c * 32 + hi * 8];
        Af.h2[0] = *(const v8h*)ap;
        Af.h2[1] = *(const v8h*)(ap + 16);
        const v8h* bp0 = (const v8h*)(wE + (size_t)((c * NTILES + t0) * 32 + lane) * 16);
        const v8h* bp1 = (const v8h*)(wE + (size_t)((c * NTILES + t1) * 32 + lane) * 16);
        B0.h2[0] = bp0[0]; B0.h2[1] = bp0[1];
        B1.h2[0] = bp1[0]; B1.h2[1] = bp1[1];
        // prefetch next chunk's fragments (L2-resident weight stream)
        if (c + 1 < KE_CHUNKS) {
            __builtin_prefetch(wE + (size_t)(((c+1) * NTILES + t0) * 32 + lane) * 16, 0, 3);
            __builtin_prefetch(wE + (size_t)(((c+1) * NTILES + t1) * 32 + lane) * 16, 0, 3);
        }
        acc0 = __builtin_amdgcn_wmma_f32_16x16x32_f16(false, Af.v, false, B0.v, (short)0, acc0, false, false);
        acc1 = __builtin_amdgcn_wmma_f32_16x16x32_f16(false, Af.v, false, B1.v, (short)0, acc1, false, false);
    }
    for (int r = 0; r < 8; ++r) {
        const int rr = hi ? (r + 8) : r;
        const int n0 = t0 * 16 + row, n1 = t1 * 16 + row;
        outT[rr][n0] = acc0[r] + bE[n0];
        outT[rr][n1] = acc1[r] + bE[n1];
    }
    __syncthreads();

    // ---- layernorm over H=128 ----------------------------------------------
    const int lrow = tid >> 3, seg = tid & 7;
    float s1 = 0.0f, s2 = 0.0f;
    for (int j = 0; j < 16; ++j) { float x = outT[lrow][seg*16 + j]; s1 += x; s2 += x*x; }
    psum[lrow][seg] = s1; psq[lrow][seg] = s2;
    __syncthreads();
    if (tid < 16) {
        float S1 = 0.0f, S2 = 0.0f;
        for (int j = 0; j < 8; ++j) { S1 += psum[tid][j]; S2 += psq[tid][j]; }
        const float mean = S1 / (float)HDIM;
        const float var  = S2 / (float)HDIM - mean * mean;
        mrow[tid] = mean;
        rrow[tid] = rsqrtf(var + 1e-05f);
    }
    __syncthreads();
    const float mean = mrow[lrow], rstd = rrow[lrow];
    const size_t eo = (size_t)blockIdx.x * 16 + lrow;
    for (int j = 0; j < 16; ++j) {
        const int n = seg * 16 + j;
        const float x = outT[lrow][n];
        outE[eo * HDIM + n] = gE[n] * (x - mean) * rstd + beE[n];
    }
}

// ---------------------------------------------------------------------------
extern "C" void kernel_launch(void* const* d_in, const int* in_sizes, int n_in,
                              void* d_out, int out_size, void* d_ws, size_t ws_size,
                              hipStream_t stream) {
    const float* coords  = (const float*)d_in[0];
    const float* central = (const float*)d_in[1];
    const int*   batch   = (const int*)d_in[2];
    const int*   chain   = (const int*)d_in[3];
    const int*   eidx    = (const int*)d_in[4];
    const float* w_node  = (const float*)d_in[5];
    const float* b_node  = (const float*)d_in[6];
    const float* g_node  = (const float*)d_in[7];
    const float* be_node = (const float*)d_in[8];
    const float* w_edge  = (const float*)d_in[9];
    const float* b_edge  = (const float*)d_in[10];
    const float* g_edge  = (const float*)d_in[11];
    const float* be_edge = (const float*)d_in[12];

    uint8_t* ws = (uint8_t*)d_ws;
    float*     Qw  = (float*)(ws + Q_OFF);
    int*       qvw = (int*)(ws + QV_OFF);
    _Float16*  wE  = (_Float16*)(ws + WE_OFF);
    _Float16*  wN  = (_Float16*)(ws + WN_OFF);
    _Float16*  hV  = (_Float16*)(ws + HV_OFF);

    float* out_hV = (float*)d_out;                       // NRES * 128
    float* out_hE = (float*)d_out + (size_t)NRES * HDIM; // NEDGE * 128

    // 1) pack weights to f16 B-fragments
    pack_w_kernel<<<(KE_CHUNKS*NTILES*32*16 + KN_CHUNKS*NTILES*32*16 + 255) / 256, 256, 0, stream>>>(
        w_edge, w_node, wE, wN);
    // 2) local frames
    prep_q_kernel<<<(NRES + 255) / 256, 256, 0, stream>>>(central, batch, chain, Qw, qvw);
    // 3) node features
    node_feat_kernel<<<(NRES + 255) / 256, 256, 0, stream>>>(coords, central, batch, chain, Qw, hV);
    // 4) node GEMM + LN
    node_gemm_kernel<<<NRES / 16, 128, 0, stream>>>(hV, wN, b_node, g_node, be_node, out_hV);
    // 5) fused edge featurize + GEMM + LN
    edge_kernel<<<NEDGE / 16, 128, 0, stream>>>(coords, eidx, Qw, qvw, wE, b_edge, g_edge, be_edge, out_hE);
}
